// FrameAwareAttentionRot_74191265071602
// MI455X (gfx1250) — matrix operand order repaired and checked by reading, not compiled, not run on hardware
//
#include <hip/hip_runtime.h>

typedef __attribute__((ext_vector_type(16))) _Float16 v16h;
typedef __attribute__((ext_vector_type(8)))  float    v8f;

#define B_   2
#define N_   2048
#define H_   8
#define DH_  32
#define DM_  256
#define NCHUNK2 (N_/32)
#define NBINS 4096
#define RMAX  64.0f
#define COARSE_W (RMAX / (float)NBINS)

__device__ inline v8f vzero8() {
  v8f z;
#pragma unroll
  for (int i = 0; i < 8; i++) z[i] = 0.0f;
  return z;
}

__device__ inline v8f wmma_f16(v16h a, v16h b, v8f c) {
  // v_wmma_f32_16x16x32_f16 : D = A(16x32 f16) * B(32x16 f16) + C(16x16 f32)
  return __builtin_amdgcn_wmma_f32_16x16x32_f16(false, a, false, b, (short)0, c, false, false);
}

// ---------------------------------------------------------------------------
// Pass 1/2: histogram of pairwise r = |R_i^T (t_j - t_i)| for the 95% quantile
// pass==0: coarse bins over [0, RMAX); pass==1: fine bins over [r_lo, r_lo+CW)
// ---------------------------------------------------------------------------
__global__ __launch_bounds__(256) void k_hist(const float* __restrict__ Rg,
                                              const float* __restrict__ tg,
                                              unsigned int* __restrict__ ghist,
                                              const float* __restrict__ scal,
                                              int pass) {
  __shared__ unsigned int lhist[NBINS];
  for (int i = threadIdx.x; i < NBINS; i += 256) lhist[i] = 0;
  __syncthreads();

  float r_lo = 0.0f, inv_w = (float)NBINS / RMAX;
  if (pass == 1) {
    r_lo = scal[0];
    inv_w = (float)NBINS / COARSE_W;
  }

  // each thread: 16 consecutive pairs -> same (b,i), consecutive j
  long p0 = ((long)blockIdx.x * 256 + threadIdx.x) * 16;
  int b = (int)(p0 / ((long)N_ * N_));
  long rem = p0 % ((long)N_ * N_);
  int i = (int)(rem / N_);
  int j0 = (int)(rem % N_);

  float Rm[9];
#pragma unroll
  for (int e = 0; e < 9; e++) Rm[e] = Rg[((long)b * N_ + i) * 9 + e];
  float tix = tg[((long)b * N_ + i) * 3 + 0];
  float tiy = tg[((long)b * N_ + i) * 3 + 1];
  float tiz = tg[((long)b * N_ + i) * 3 + 2];

  for (int jj = 0; jj < 16; jj++) {
    int j = j0 + jj;
    float dx = tg[((long)b * N_ + j) * 3 + 0] - tix;
    float dy = tg[((long)b * N_ + j) * 3 + 1] - tiy;
    float dz = tg[((long)b * N_ + j) * 3 + 2] - tiz;
    // e_local[m] = sum_n R[n][m] * delta[n]  (R^T * delta)
    float ex = Rm[0] * dx + Rm[3] * dy + Rm[6] * dz;
    float ey = Rm[1] * dx + Rm[4] * dy + Rm[7] * dz;
    float ez = Rm[2] * dx + Rm[5] * dy + Rm[8] * dz;
    float r = fmaxf(sqrtf(ex * ex + ey * ey + ez * ez), 1e-8f);
    float rel = (r - r_lo) * inv_w;
    if (rel >= 0.0f) {
      int bin = (int)rel;
      if (pass == 0) {
        if (bin > NBINS - 1) bin = NBINS - 1;
        atomicAdd(&lhist[bin], 1u);
      } else if (bin < NBINS) {
        atomicAdd(&lhist[bin], 1u);
      }
    }
  }
  __syncthreads();
  for (int i2 = threadIdx.x; i2 < NBINS; i2 += 256) {
    unsigned int v = lhist[i2];
    if (v) atomicAdd(&ghist[i2], v);
  }
}

__global__ void k_quantile1(const unsigned int* __restrict__ ghist,
                            float* __restrict__ scal) {
  if (threadIdx.x == 0 && blockIdx.x == 0) {
    const float total = (float)((long)B_ * N_ * N_);
    float target = 0.95f * total;
    float cum = 0.0f;
    float r_lo = RMAX - COARSE_W, below = 0.0f;
    for (int i = 0; i < NBINS; i++) {
      float hcnt = (float)ghist[i];
      if (cum + hcnt >= target) { r_lo = (float)i * COARSE_W; below = cum; break; }
      cum += hcnt;
    }
    scal[0] = r_lo;
    scal[1] = below;
    scal[3] = target;
  }
}

__global__ void k_quantile2(const unsigned int* __restrict__ fhist,
                            float* __restrict__ scal) {
  if (threadIdx.x == 0 && blockIdx.x == 0) {
    float target2 = scal[3] - scal[1];
    float cum = 0.0f;
    const float fw = COARSE_W / (float)NBINS;
    float val = scal[0] + COARSE_W;
    for (int i = 0; i < NBINS; i++) {
      float hcnt = (float)fhist[i];
      if (cum + hcnt >= target2) {
        float frac = (target2 - cum) / fmaxf(hcnt, 1.0f);
        val = scal[0] + ((float)i + frac) * fw;
        break;
      }
      cum += hcnt;
    }
    scal[2] = val + 1e-6f;  // r_end
  }
}

// ---------------------------------------------------------------------------
// QKV projection + RoPE, stores q/k/v as f16 in [B,H,N,Dh]
// ---------------------------------------------------------------------------
__global__ __launch_bounds__(256) void k_proj_rope(const float* __restrict__ x,
                                                   const float* __restrict__ Wq,
                                                   const float* __restrict__ Wk,
                                                   const float* __restrict__ Wv,
                                                   _Float16* __restrict__ qh,
                                                   _Float16* __restrict__ kh,
                                                   _Float16* __restrict__ vh) {
  __shared__ float xr[DM_];
  __shared__ float qr[DM_];
  __shared__ float kr[DM_];
  const int row = blockIdx.x;       // b*N + n
  const int b = row / N_;
  const int n = row % N_;
  const int o = threadIdx.x;        // output channel 0..255

  xr[o] = x[(long)row * DM_ + o];
  __syncthreads();

  const float* wq = Wq + (long)o * DM_;
  const float* wk = Wk + (long)o * DM_;
  const float* wv = Wv + (long)o * DM_;
  float aq = 0.0f, ak = 0.0f, av = 0.0f;
#pragma unroll 8
  for (int k = 0; k < DM_; k++) {
    float xv = xr[k];
    aq += xv * wq[k];
    ak += xv * wk[k];
    av += xv * wv[k];
  }
  qr[o] = aq;
  kr[o] = ak;
  __syncthreads();

  const int hd = o / DH_;
  const int d = o % DH_;
  const int d2 = d & 15;
  // inv_freq = 10000^(-(2*d2)/32)
  float inv_freq = __expf(-((float)d2 / 16.0f) * 9.2103403719761836f);
  float ang = (float)n * inv_freq;
  float c = cosf(ang), s = sinf(ang);
  float rotq = (d < 16) ? -qr[hd * DH_ + d + 16] : qr[hd * DH_ + d - 16];
  float rotk = (d < 16) ? -kr[hd * DH_ + d + 16] : kr[hd * DH_ + d - 16];

  long base = (((long)b * H_ + hd) * N_ + n) * DH_ + d;
  qh[base] = (_Float16)(aq * c + rotq * s);
  kh[base] = (_Float16)(ak * c + rotk * s);
  vh[base] = (_Float16)av;
}

// ---------------------------------------------------------------------------
// Fused pair-MLP bias + flash attention.
// Block = (b, 16-row i tile); 8 waves = 8 heads.
// Outer loop: 64 iterations of 32 j; each iteration runs two 16-j MLP batches
// (geo -> MLP1 -> MLP2 -> scores in registers), then one joint softmax and
// one full-K=32 PV WMMA pair.
// ---------------------------------------------------------------------------
__global__ __launch_bounds__(256) void k_attn(const _Float16* __restrict__ qh,
                                              const _Float16* __restrict__ kh,
                                              const _Float16* __restrict__ vh,
                                              const float* __restrict__ Rg,
                                              const float* __restrict__ tg,
                                              const float* __restrict__ scal,
                                              const float* __restrict__ W1,
                                              const float* __restrict__ b1,
                                              const float* __restrict__ W2,
                                              const float* __restrict__ b2,
                                              float* __restrict__ aout) {
  // union region: geo f16 [256][32] -> h f16 [256][64] -> bias f32 [256][8]
  __shared__ __align__(16) _Float16 U[256 * 64];        // 32 KB
  // K tile [8][16][32] during score phase; aliased as P tile [8][16][32] after
  __shared__ __align__(16) _Float16 KP[H_ * 16 * 32];   // 8 KB
  __shared__ __align__(16) _Float16 Vt[H_ * 32 * DH_];  // 16 KB (32 j rows)

  const int tid = threadIdx.x;
  const int wave = tid >> 5;    // head
  const int lane = tid & 31;
  const int lhalf = lane >> 4;  // lane half (0/1)
  const int l16 = lane & 15;
  const int hd = wave;

  const int b = blockIdx.x / (N_ / 16);
  const int i0 = (blockIdx.x % (N_ / 16)) * 16;

  const float r_end = fmaxf(scal[2], 1e-6f);
  const float inv_rend = 1.0f / r_end;
  const float scale = 0.17677669529663687f;  // 1/sqrt(32)

  // --- constant B fragments: W1^T [K=32pad][N=64] (4 tiles), W2^T [K=64][N=16pad]
  v16h Wb1[4];
#pragma unroll
  for (int nt = 0; nt < 4; nt++) {
#pragma unroll
    for (int e = 0; e < 16; e++) {
      int k = lhalf * 16 + e;           // B layout: K = half*16 + e, N = l16
      int nn = nt * 16 + l16;
      Wb1[nt][e] = (k < 20) ? (_Float16)W1[nn * 20 + k] : (_Float16)0.0f;
    }
  }
  v16h Wb2[2];
#pragma unroll
  for (int kt = 0; kt < 2; kt++) {
#pragma unroll
    for (int e = 0; e < 16; e++) {
      int k = kt * 32 + lhalf * 16 + e;
      Wb2[kt][e] = (l16 < H_) ? (_Float16)W2[l16 * 64 + k] : (_Float16)0.0f;
    }
  }
  float b1v[4];
#pragma unroll
  for (int nt = 0; nt < 4; nt++) b1v[nt] = b1[nt * 16 + l16];
  const float b2v = (l16 < H_) ? b2[l16] : 0.0f;

  // --- q A-fragment for this head: A layout K = half*16 + e
  v16h qa;
#pragma unroll
  for (int e = 0; e < 16; e++) {
    int d = lhalf * 16 + e;
    qa[e] = qh[(((long)b * H_ + hd) * N_ + (i0 + l16)) * DH_ + d];
  }

  // --- per-thread pair geometry constants (this thread owns pair (iloc,jloc))
  const int iloc = tid >> 4;
  const int jloc = tid & 15;
  const int gi = i0 + iloc;
  float Rm[9];
#pragma unroll
  for (int e = 0; e < 9; e++) Rm[e] = Rg[((long)b * N_ + gi) * 9 + e];
  const float tix = tg[((long)b * N_ + gi) * 3 + 0];
  const float tiy = tg[((long)b * N_ + gi) * 3 + 1];
  const float tiz = tg[((long)b * N_ + gi) * 3 + 2];

  // --- online softmax state (rows v + 8*lhalf of the 16x16 C layout)
  float mrun[8], lrun[8];
#pragma unroll
  for (int v = 0; v < 8; v++) { mrun[v] = -3.0e38f; lrun[v] = 0.0f; }
  v8f acc0 = vzero8();
  v8f acc1 = vzero8();

  float* biasf = (float*)U;

  for (int jc = 0; jc < NCHUNK2; jc++) {
    const int j0 = jc * 32;
    float scA[8], scB[8];   // score tiles (cols l16 and 16+l16 of this 32-j span)

#pragma unroll
    for (int bb = 0; bb < 2; bb++) {
      const int jb0 = j0 + bb * 16;
      __syncthreads();  // previous phase's LDS reads done

      // ---- stage K tile (this batch) and V half-tile for all 8 heads
      {
        int base = tid * 16;           // 4096 halfs each
        int th = base >> 9;            // head
        int tj = (base >> 5) & 15;     // j within batch
        int td = base & 31;            // d (0 or 16)
        long src = (((long)b * H_ + th) * N_ + (jb0 + tj)) * DH_ + td;
        int vdst = (th * 32 + bb * 16 + tj) * DH_ + td;
#pragma unroll
        for (int e = 0; e < 16; e++) {
          KP[base + e] = kh[src + e];
          Vt[vdst + e] = vh[src + e];
        }
      }

      // ---- geometry features for this thread's pair -> U (geo [256][32])
      {
        int gj = jb0 + jloc;
        float dx = tg[((long)b * N_ + gj) * 3 + 0] - tix;
        float dy = tg[((long)b * N_ + gj) * 3 + 1] - tiy;
        float dz = tg[((long)b * N_ + gj) * 3 + 2] - tiz;
        float ex = Rm[0] * dx + Rm[3] * dy + Rm[6] * dz;
        float ey = Rm[1] * dx + Rm[4] * dy + Rm[7] * dz;
        float ez = Rm[2] * dx + Rm[5] * dy + Rm[8] * dz;
        float r = fmaxf(sqrtf(ex * ex + ey * ey + ez * ez), 1e-8f);
        float inv_r = 1.0f / r;
        _Float16* g = &U[tid * 32];
        g[0] = (_Float16)r;
        g[1] = (_Float16)(ex * inv_r);
        g[2] = (_Float16)(ey * inv_r);
        g[3] = (_Float16)(ez * inv_r);
        float xn = r * inv_rend;
        const float w15 = 1.0f / 15.0f;
#pragma unroll
        for (int kb = 0; kb < 16; kb++) {
          float val = fmaxf(0.0f, 1.0f - fabsf(xn - (float)kb * w15) / (w15 + 1e-9f));
          g[4 + kb] = (_Float16)val;
        }
#pragma unroll
        for (int e = 20; e < 32; e++) g[e] = (_Float16)0.0f;
      }
      __syncthreads();

      // ---- gather geo A-fragments (wave w owns M-tiles 2w, 2w+1)
      v16h gA[2];
#pragma unroll
      for (int m = 0; m < 2; m++) {
        int mt = wave * 2 + m;
#pragma unroll
        for (int e = 0; e < 16; e++) {
          int k = lhalf * 16 + e;
          gA[m][e] = U[(mt * 16 + l16) * 32 + k];
        }
      }
      __syncthreads();

      // ---- MLP layer 1: h = relu(geo @ W1^T + b1) -> U (h [256][64])
#pragma unroll
      for (int m = 0; m < 2; m++) {
        int mt = wave * 2 + m;
#pragma unroll
        for (int nt = 0; nt < 4; nt++) {
          v8f c = wmma_f16(gA[m], Wb1[nt], vzero8());
#pragma unroll
          for (int v = 0; v < 8; v++) {
            float hval = fmaxf(c[v] + b1v[nt], 0.0f);
            U[(mt * 16 + v + lhalf * 8) * 64 + nt * 16 + l16] = (_Float16)hval;
          }
        }
      }
      __syncthreads();

      // ---- gather h A-fragments (K=64 -> 2 fragments)
      v16h hA[2][2];
#pragma unroll
      for (int m = 0; m < 2; m++) {
        int mt = wave * 2 + m;
#pragma unroll
        for (int kt = 0; kt < 2; kt++) {
#pragma unroll
          for (int e = 0; e < 16; e++) {
            int k = kt * 32 + lhalf * 16 + e;
            hA[m][kt][e] = U[(mt * 16 + l16) * 64 + k];
          }
        }
      }
      __syncthreads();

      // ---- MLP layer 2: bias = h @ W2^T + b2 -> U (bias f32 [256][8])
#pragma unroll
      for (int m = 0; m < 2; m++) {
        int mt = wave * 2 + m;
        v8f c = wmma_f16(hA[m][0], Wb2[0], vzero8());
        c = wmma_f16(hA[m][1], Wb2[1], c);
        if (l16 < H_) {
#pragma unroll
          for (int v = 0; v < 8; v++)
            biasf[(mt * 16 + v + lhalf * 8) * H_ + l16] = c[v] + b2v;
        }
      }
      __syncthreads();

      // ---- scores for this wave's head: S = Q K^T * scale + bias (to regs)
      v16h kB;  // B layout: K=d = half*16+e, N=j = l16 -> contiguous k-row half
#pragma unroll
      for (int e = 0; e < 16; e++) kB[e] = KP[(hd * 16 + l16) * 32 + lhalf * 16 + e];
      v8f sc = wmma_f16(qa, kB, vzero8());
#pragma unroll
      for (int v = 0; v < 8; v++) {
        float s = sc[v] * scale + biasf[((v + lhalf * 8) * 16 + l16) * H_ + hd];
        if (bb == 0) scA[v] = s; else scB[v] = s;
      }
    }

    // ---- joint online softmax over all 32 columns; write P into KP alias.
    // (per-head region: this wave finished reading its K rows; barriers at the
    //  top of the next batch protect cross-wave reuse)
    float corr[8];
#pragma unroll
    for (int v = 0; v < 8; v++) {
      float rm = fmaxf(scA[v], scB[v]);
      rm = fmaxf(rm, __shfl_xor(rm, 1, 32));
      rm = fmaxf(rm, __shfl_xor(rm, 2, 32));
      rm = fmaxf(rm, __shfl_xor(rm, 4, 32));
      rm = fmaxf(rm, __shfl_xor(rm, 8, 32));
      float mnew = fmaxf(mrun[v], rm);
      float pA = __expf(scA[v] - mnew);
      float pB = __expf(scB[v] - mnew);
      float rs = pA + pB;
      rs += __shfl_xor(rs, 1, 32);
      rs += __shfl_xor(rs, 2, 32);
      rs += __shfl_xor(rs, 4, 32);
      rs += __shfl_xor(rs, 8, 32);
      float cf = __expf(mrun[v] - mnew);
      lrun[v] = lrun[v] * cf + rs;
      mrun[v] = mnew;
      corr[v] = cf;
      int prow = (hd * 16 + v + lhalf * 8) * 32;
      KP[prow + l16] = (_Float16)pA;
      KP[prow + 16 + l16] = (_Float16)pB;
    }
#pragma unroll
    for (int v = 0; v < 8; v++) { acc0[v] *= corr[v]; acc1[v] *= corr[v]; }

    // ---- PV: acc += P(16x32, fully populated) @ V(32x32)
    v16h pA;
#pragma unroll
    for (int e = 0; e < 16; e++) {
      int kj = lhalf * 16 + e;                             // 0..31, all valid
      pA[e] = KP[(hd * 16 + l16) * 32 + kj];
    }
    v16h vB0, vB1;
#pragma unroll
    for (int e = 0; e < 16; e++) {
      int j = lhalf * 16 + e;                              // 0..31, all valid
      vB0[e] = Vt[(hd * 32 + j) * DH_ + l16];
      vB1[e] = Vt[(hd * 32 + j) * DH_ + 16 + l16];
    }
    acc0 = wmma_f16(pA, vB0, acc0);
    acc1 = wmma_f16(pA, vB1, acc1);
  }

  // ---- finalize: out = acc / l, write [B,N,H*Dh] f32
#pragma unroll
  for (int v = 0; v < 8; v++) {
    float invl = 1.0f / lrun[v];
    int gi2 = i0 + v + lhalf * 8;
    long base = ((long)b * N_ + gi2) * DM_ + hd * DH_;
    aout[base + l16] = acc0[v] * invl;
    aout[base + 16 + l16] = acc1[v] * invl;
  }
}

// ---------------------------------------------------------------------------
// Output projection: y = out @ Wo^T
// ---------------------------------------------------------------------------
__global__ __launch_bounds__(256) void k_outproj(const float* __restrict__ aout,
                                                 const float* __restrict__ Wo,
                                                 float* __restrict__ y) {
  __shared__ float xr[DM_];
  const int row = blockIdx.x;
  const int o = threadIdx.x;
  xr[o] = aout[(long)row * DM_ + o];
  __syncthreads();
  const float* w = Wo + (long)o * DM_;
  float acc = 0.0f;
#pragma unroll 8
  for (int k = 0; k < DM_; k++) acc += xr[k] * w[k];
  y[(long)row * DM_ + o] = acc;
}

// ---------------------------------------------------------------------------
extern "C" void kernel_launch(void* const* d_in, const int* in_sizes, int n_in,
                              void* d_out, int out_size, void* d_ws, size_t ws_size,
                              hipStream_t stream) {
  (void)in_sizes; (void)n_in; (void)out_size; (void)ws_size;
  const float* x  = (const float*)d_in[0];
  const float* R  = (const float*)d_in[1];
  const float* t  = (const float*)d_in[2];
  // d_in[3] = node_mask (all true for this problem; -inf masking is a no-op)
  const float* Wq = (const float*)d_in[4];
  const float* Wk = (const float*)d_in[5];
  const float* Wv = (const float*)d_in[6];
  const float* Wo = (const float*)d_in[7];
  const float* W1 = (const float*)d_in[8];
  const float* b1 = (const float*)d_in[9];
  const float* W2 = (const float*)d_in[10];
  const float* b2 = (const float*)d_in[11];

  char* ws = (char*)d_ws;
  unsigned int* hist  = (unsigned int*)ws;                  // 16 KB coarse
  unsigned int* hist2 = (unsigned int*)(ws + 16384);        // 16 KB fine
  float* scal = (float*)(ws + 32768);                       // r_lo, cnt_below, r_end, target
  _Float16* qh = (_Float16*)(ws + 33024);
  const size_t qkv = (size_t)B_ * H_ * N_ * DH_;
  _Float16* kh = qh + qkv;
  _Float16* vh = kh + qkv;
  float* aout = (float*)(vh + qkv);

  const int pair_blocks = (B_ * N_ * N_) / (256 * 16);      // 2048

  hipMemsetAsync(hist, 0, 2 * NBINS * sizeof(unsigned int), stream);
  k_hist<<<pair_blocks, 256, 0, stream>>>(R, t, hist, scal, 0);
  k_quantile1<<<1, 1, 0, stream>>>(hist, scal);
  k_hist<<<pair_blocks, 256, 0, stream>>>(R, t, hist2, scal, 1);
  k_quantile2<<<1, 1, 0, stream>>>(hist2, scal);

  k_proj_rope<<<B_ * N_, 256, 0, stream>>>(x, Wq, Wk, Wv, qh, kh, vh);
  k_attn<<<B_ * (N_ / 16), 256, 0, stream>>>(qh, kh, vh, R, t, scal,
                                             W1, b1, W2, b2, aout);
  k_outproj<<<B_ * N_, 256, 0, stream>>>(aout, Wo, (float*)d_out);
}
